// MoEAttention_7267084665019
// MI455X (gfx1250) — matrix operand order repaired and checked by reading, not compiled, not run on hardware
//
#include <hip/hip_runtime.h>
#include <hip/hip_bf16.h>

// ---------------------------------------------------------------------------
// Problem constants (fixed by the reference): B=2, S=2048, H=2048, 16 x 128
// ---------------------------------------------------------------------------
#define BATCH   2
#define SEQ     2048
#define HID     2048
#define NHEADS  16
#define HDIM    128
#define MTOK    (BATCH * SEQ)          // 4096 rows in all projections

typedef __bf16 bf16_t;
typedef __attribute__((ext_vector_type(8)))  __bf16 v8bf;
typedef __attribute__((ext_vector_type(16))) __bf16 v16bf;
typedef __attribute__((ext_vector_type(8)))  float  v8f;
typedef __attribute__((ext_vector_type(4)))  int    v4i;

#if __has_builtin(__builtin_amdgcn_global_load_async_to_lds_b128) && \
    __has_builtin(__builtin_amdgcn_s_wait_asynccnt)
#define HAS_ASYNC_LDS 1
#else
#define HAS_ASYNC_LDS 0
#endif

__device__ __forceinline__ bf16_t f2bf(float x) {
  unsigned u = __builtin_bit_cast(unsigned, x);
  u += 0x7FFFu + ((u >> 16) & 1u);               // round-to-nearest-even
  unsigned short h = (unsigned short)(u >> 16);
  return __builtin_bit_cast(bf16_t, h);
}
__device__ __forceinline__ float bf2f(bf16_t x) {
  unsigned short h = __builtin_bit_cast(unsigned short, x);
  unsigned u = ((unsigned)h) << 16;
  return __builtin_bit_cast(float, u);
}
__device__ __forceinline__ v16bf cat16(v8bf lo, v8bf hi) {
  return __builtin_shufflevector(lo, hi, 0,1,2,3,4,5,6,7,8,9,10,11,12,13,14,15);
}

// 16-byte global -> LDS copy: async DMA on CDNA5 (ASYNCcnt), else via VGPRs.
// The gfx1250 builtin takes (v4i addrspace(1)*, v4i addrspace(3)*, imm, cpol).
__device__ __forceinline__ void cp16_g2l(const bf16_t* g, bf16_t* l) {
#if HAS_ASYNC_LDS
  __builtin_amdgcn_global_load_async_to_lds_b128(
      (__attribute__((address_space(1))) v4i*)(bf16_t*)g,
      (__attribute__((address_space(3))) v4i*)l, 0, 0);
#else
  *(v8bf*)l = *(const v8bf*)g;
#endif
}

// ---------------------------------------------------------------------------
// fp32 -> bf16 bulk convert (4 elems/thread, n multiple of 1024)
// ---------------------------------------------------------------------------
__global__ void cvt_f32_bf16(const float* __restrict__ src,
                             bf16_t* __restrict__ dst, int n) {
  int i = (blockIdx.x * blockDim.x + threadIdx.x) * 4;
  if (i + 3 < n) {
    float4 v = *(const float4*)(src + i);
    dst[i + 0] = f2bf(v.x);
    dst[i + 1] = f2bf(v.y);
    dst[i + 2] = f2bf(v.z);
    dst[i + 3] = f2bf(v.w);
  }
}

// ---------------------------------------------------------------------------
// RoPE cos/sin table: [SEQ][HDIM], duplicated halves like the reference
// ---------------------------------------------------------------------------
__global__ void rope_table(float* __restrict__ cosT, float* __restrict__ sinT) {
  int s = blockIdx.x;
  int d = threadIdx.x;                 // 0..127
  int dd = d & 63;
  float inv = __expf(-((float)(2 * dd) / (float)HDIM) * __logf(10000.0f));
  float fr = (float)s * inv;
  cosT[s * HDIM + d] = __cosf(fr);
  sinT[s * HDIM + d] = __sinf(fr);
}

// In-place RoPE on bf16 tensor laid out [B, nH, S, hd]; 64 threads handle a
// (d, d+64) pair so there is no intra-block RAW hazard.
__global__ void rope_apply(bf16_t* __restrict__ q,
                           const float* __restrict__ cosT,
                           const float* __restrict__ sinT) {
  int bhs = blockIdx.x;                // b*nH*S + h*S + s
  int s = bhs & (SEQ - 1);
  int d = threadIdx.x;                 // 0..63
  bf16_t* p = q + (size_t)bhs * HDIM;
  float x1 = bf2f(p[d]);
  float x2 = bf2f(p[d + 64]);
  float c  = cosT[s * HDIM + d];
  float sn = sinT[s * HDIM + d];
  p[d]      = f2bf(x1 * c - x2 * sn);
  p[d + 64] = f2bf(x2 * c + x1 * sn);
}

// ---------------------------------------------------------------------------
// Tiled bf16 WMMA GEMM:  C[m,n] = sum_k A[m,k] * W[n,k]   (torch Linear)
// Block tile 128x128, 8 waves (2x4), per-wave 64x32 via 4x2 accumulators.
// K staged in 64-deep slabs through double-buffered LDS that is filled with
// GLOBAL_LOAD_ASYNC_TO_LDS_B128 (ASYNCcnt): the next slab's DMA overlaps the
// current slab's 16 WMMAs. Fragment reads follow the CDNA5 16-bit A/B VGPR
// layouts (ISA 7.12.2): every lane does aligned 16B LDS reads.
// OUT_MODE: 0 -> bf16 [B,nH,S,hd] (Q,K)    1 -> bf16 [B,nH,hd,S] (V^T)
//           2 -> f32  [M,N] row-major (final output)
// ---------------------------------------------------------------------------
template <int OUT_MODE>
__global__ __launch_bounds__(256)
void gemm_bf16_wmma(const bf16_t* __restrict__ A,
                    const bf16_t* __restrict__ W,
                    void* __restrict__ outp) {
  constexpr int K = HID, N = HID;
  constexpr int BK = 64;
  __shared__ bf16_t sA[2][128 * BK];   // 2 x 16 KB
  __shared__ bf16_t sW[2][128 * BK];   // 2 x 16 KB

  const int tid  = threadIdx.x;
  const int lane = tid & 31;
  const int wave = tid >> 5;
  const int wm   = wave >> 2;          // 0..1
  const int wn   = wave & 3;           // 0..3
  const int m0   = blockIdx.y * 128;
  const int n0   = blockIdx.x * 128;
  const int half = lane >> 4;
  const int l16  = lane & 15;

  v8f zero = {};
  v8f acc[4][2];
#pragma unroll
  for (int i = 0; i < 4; ++i)
#pragma unroll
    for (int j = 0; j < 2; ++j) acc[i][j] = zero;

  // Staging: 128 rows x 64 cols per operand; 2 threads per row, 32 elems each.
  const int ldRow = tid >> 1;          // 0..127
  const int ldCol = (tid & 1) * 32;    // 0 / 32

  const bf16_t* gA = A + (size_t)(m0 + ldRow) * K + ldCol;
  const bf16_t* gW = W + (size_t)(n0 + ldRow) * K + ldCol;

  auto stage = [&](int kt, int buf) {
    bf16_t* la = &sA[buf][ldRow * BK + ldCol];
    bf16_t* lw = &sW[buf][ldRow * BK + ldCol];
#pragma unroll
    for (int c = 0; c < 4; ++c) {      // 4 x 16B per operand
      cp16_g2l(gA + kt + c * 8, la + c * 8);
      cp16_g2l(gW + kt + c * 8, lw + c * 8);
    }
  };

  stage(0, 0);
  int buf = 0;
  for (int kt = 0; kt < K; kt += BK) {
    if (kt + BK < K) {
      stage(kt + BK, buf ^ 1);         // overlap next slab's DMA with compute
#if HAS_ASYNC_LDS
      __builtin_amdgcn_s_wait_asynccnt(8);   // this slab's 8 oldest landed
#endif
    } else {
#if HAS_ASYNC_LDS
      __builtin_amdgcn_s_wait_asynccnt(0);
#endif
    }
    __syncthreads();

#pragma unroll
    for (int kc = 0; kc < 2; ++kc) {   // two 32-deep WMMA steps per slab
      v16bf aF[4], bF[2];
#pragma unroll
      for (int mf = 0; mf < 4; ++mf) {
        const bf16_t* pr = &sA[buf][(wm * 64 + mf * 16 + l16) * BK + kc * 32];
        v8bf lo = *(const v8bf*)(pr + half * 8);        // K = half*8 + 0..7
        v8bf hi = *(const v8bf*)(pr + 16 + half * 8);   // K = 16+half*8+0..7
        aF[mf] = cat16(lo, hi);
      }
#pragma unroll
      for (int nf = 0; nf < 2; ++nf) {
        const bf16_t* pr =
            &sW[buf][(wn * 32 + nf * 16 + l16) * BK + kc * 32 + half * 8];
        v8bf lo = *(const v8bf*)(pr);                   // K = half*8 + 0..7
        v8bf hi = *(const v8bf*)(pr + 16);              // K = 16+half*8+0..7
        bF[nf] = cat16(lo, hi);
      }
#pragma unroll
      for (int mf = 0; mf < 4; ++mf)
#pragma unroll
        for (int nf = 0; nf < 2; ++nf)
          acc[mf][nf] = __builtin_amdgcn_wmma_f32_16x16x32_bf16(
              false, aF[mf], false, bF[nf], (short)0, acc[mf][nf],
              false, false);
    }
    __syncthreads();
    buf ^= 1;
  }

  // C/D layout: VGPR r, lanes 0-15 -> M=r, lanes 16-31 -> M=r+8, N=lane&15
#pragma unroll
  for (int mf = 0; mf < 4; ++mf) {
#pragma unroll
    for (int nf = 0; nf < 2; ++nf) {
#pragma unroll
      for (int r = 0; r < 8; ++r) {
        int m = m0 + wm * 64 + mf * 16 + r + half * 8;
        int n = n0 + wn * 32 + nf * 16 + l16;
        float v = acc[mf][nf][r];
        if constexpr (OUT_MODE == 2) {
          ((float*)outp)[(size_t)m * N + n] = v;
        } else {
          int b = m >> 11, s = m & (SEQ - 1);
          int h = n >> 7,  d = n & (HDIM - 1);
          bf16_t* o = (bf16_t*)outp;
          if constexpr (OUT_MODE == 0)
            o[(((size_t)(b * NHEADS + h)) * SEQ + s) * HDIM + d] = f2bf(v);
          else
            o[(((size_t)(b * NHEADS + h)) * HDIM + d) * SEQ + s] = f2bf(v);
        }
      }
    }
  }
}

// ---------------------------------------------------------------------------
// Fused causal flash attention, bf16 WMMA, f32 online softmax.
// Grid: (B*nH, S/64); block = 128 threads = 4 waves; each wave owns 16
// queries and the full 128-wide output. Per 32-key tile: 8 WMMAs for Q@K^T,
// half-wave shfl_xor row reductions, LDS re-layout of P (C-layout -> A-layout),
// 8 WMMAs for P@V against the transposed V buffer.
// ---------------------------------------------------------------------------
__global__ __launch_bounds__(128)
void flash_attn_wmma(const bf16_t* __restrict__ Qb,
                     const bf16_t* __restrict__ Kb,
                     const bf16_t* __restrict__ VTb,
                     bf16_t* __restrict__ Ab) {
  constexpr float SCALE = 0.08838834764831845f;   // 1/sqrt(128)
  __shared__ bf16_t sP[4][16 * 32];

  const int bh   = blockIdx.x;                    // 0..31
  const int b    = bh >> 4, h = bh & (NHEADS - 1);
  const int lane = threadIdx.x & 31;
  const int wave = threadIdx.x >> 5;
  const int half = lane >> 4;
  const int l16  = lane & 15;
  const int q0   = blockIdx.y * 64 + wave * 16;   // first query row of wave

  const bf16_t* Qh = Qb  + (size_t)bh * SEQ * HDIM;
  const bf16_t* Kh = Kb  + (size_t)bh * SEQ * HDIM;
  const bf16_t* Vt = VTb + (size_t)bh * HDIM * SEQ;
  bf16_t* myP = sP[wave];

  // Q fragments: 16 rows x 128 dims as 4 chunks of K=32 (A layout)
  v16bf qf[4];
#pragma unroll
  for (int c = 0; c < 4; ++c) {
    const bf16_t* pr = Qh + (size_t)(q0 + l16) * HDIM + c * 32 + half * 8;
    qf[c] = cat16(*(const v8bf*)pr, *(const v8bf*)(pr + 16));
  }

  v8f zero = {};
  v8f acc[8];
#pragma unroll
  for (int i = 0; i < 8; ++i) acc[i] = zero;
  float mrow[8], lrow[8];
#pragma unroll
  for (int r = 0; r < 8; ++r) { mrow[r] = -1e30f; lrow[r] = 0.0f; }

  for (int kt = 0; kt < q0 + 16; kt += 32) {
    // ---- scores: S = Q @ K^T over this 32-key tile ----
    v8f s0 = zero, s1 = zero;
#pragma unroll
    for (int c = 0; c < 4; ++c) {
      const bf16_t* pk0 = Kh + (size_t)(kt + l16) * HDIM + c * 32 + half * 8;
      const bf16_t* pk1 = Kh + (size_t)(kt + 16 + l16) * HDIM + c * 32 + half * 8;
      v16bf kF0 = cat16(*(const v8bf*)pk0, *(const v8bf*)(pk0 + 16));
      v16bf kF1 = cat16(*(const v8bf*)pk1, *(const v8bf*)(pk1 + 16));
      s0 = __builtin_amdgcn_wmma_f32_16x16x32_bf16(false, qf[c], false, kF0,
                                                   (short)0, s0, false, false);
      s1 = __builtin_amdgcn_wmma_f32_16x16x32_bf16(false, qf[c], false, kF1,
                                                   (short)0, s1, false, false);
    }

    // ---- online softmax (rows live on 16-lane half-waves) ----
    float mnew[8];
#pragma unroll
    for (int r = 0; r < 8; ++r) {
      int row = q0 + r + half * 8;
      float v0 = s0[r] * SCALE; if (kt + l16 > row)      v0 = -1e30f;
      float v1 = s1[r] * SCALE; if (kt + 16 + l16 > row) v1 = -1e30f;
      s0[r] = v0; s1[r] = v1;
      float mx = fmaxf(v0, v1);
#pragma unroll
      for (int off = 1; off < 16; off <<= 1)
        mx = fmaxf(mx, __shfl_xor(mx, off, 32));
      mnew[r] = fmaxf(mrow[r], mx);
    }
#pragma unroll
    for (int r = 0; r < 8; ++r) {
      float p0 = __expf(s0[r] - mnew[r]);
      float p1 = __expf(s1[r] - mnew[r]);
      myP[(r + half * 8) * 32 + l16]      = f2bf(p0);
      myP[(r + half * 8) * 32 + 16 + l16] = f2bf(p1);
      float rs = p0 + p1;
#pragma unroll
      for (int off = 1; off < 16; off <<= 1)
        rs += __shfl_xor(rs, off, 32);
      float esc = __expf(mrow[r] - mnew[r]);
      lrow[r] = lrow[r] * esc + rs;
      mrow[r] = mnew[r];
#pragma unroll
      for (int j = 0; j < 8; ++j) acc[j][r] *= esc;
    }

    // ---- re-load P as an A-layout fragment (per-wave LDS, in-order DS) ----
    const bf16_t* pp = &myP[l16 * 32 + half * 8];
    v16bf pA = cat16(*(const v8bf*)pp, *(const v8bf*)(pp + 16));

    // ---- O += P @ V  (V^T gives contiguous B fragments) ----
#pragma unroll
    for (int nt = 0; nt < 8; ++nt) {
      const bf16_t* pv = Vt + (size_t)(nt * 16 + l16) * SEQ + kt + half * 8;
      v16bf vF = cat16(*(const v8bf*)pv, *(const v8bf*)(pv + 16));
      acc[nt] = __builtin_amdgcn_wmma_f32_16x16x32_bf16(false, pA, false, vF,
                                                        (short)0, acc[nt],
                                                        false, false);
    }
  }

  // ---- normalize and store bf16 [B,S,H] (A operand of the Wo projection) ----
#pragma unroll
  for (int r = 0; r < 8; ++r) {
    float inv = 1.0f / lrow[r];
    int row = q0 + r + half * 8;
    bf16_t* orow = Ab + ((size_t)b * SEQ + row) * HID + h * HDIM;
#pragma unroll
    for (int nt = 0; nt < 8; ++nt)
      orow[nt * 16 + l16] = f2bf(acc[nt][r] * inv);
  }
}

// ---------------------------------------------------------------------------
// Orchestration
// ---------------------------------------------------------------------------
extern "C" void kernel_launch(void* const* d_in, const int* in_sizes, int n_in,
                              void* d_out, int out_size, void* d_ws,
                              size_t ws_size, hipStream_t stream) {
  (void)in_sizes; (void)n_in; (void)out_size; (void)ws_size;

  const float* X  = (const float*)d_in[0];   // hidden_states [B,S,H]
  // d_in[1] = attention_mask (causal, reconstructed analytically)
  // d_in[2] = position_ids   (arange, reconstructed analytically)
  const float* Wq = (const float*)d_in[3];
  const float* Wk = (const float*)d_in[4];
  const float* Wv = (const float*)d_in[5];
  const float* Wo = (const float*)d_in[6];

  char* ws = (char*)d_ws;
  bf16_t* Xb   = (bf16_t*)(ws);
  bf16_t* Wqb  = (bf16_t*)(ws + ((size_t)16 << 20));
  bf16_t* Wkb  = (bf16_t*)(ws + ((size_t)24 << 20));
  bf16_t* Wvb  = (bf16_t*)(ws + ((size_t)32 << 20));
  bf16_t* Wob  = (bf16_t*)(ws + ((size_t)40 << 20));
  bf16_t* Qb   = (bf16_t*)(ws + ((size_t)48 << 20));   // [B,nH,S,hd]
  bf16_t* Kb   = (bf16_t*)(ws + ((size_t)64 << 20));   // [B,nH,S,hd]
  bf16_t* VTb  = (bf16_t*)(ws + ((size_t)80 << 20));   // [B,nH,hd,S]
  bf16_t* Ab   = (bf16_t*)(ws + ((size_t)96 << 20));   // [B,S,H]
  float*  cosT = (float*) (ws + ((size_t)112 << 20));
  float*  sinT = (float*) (ws + ((size_t)113 << 20));

  // 1) bf16 conversion of activations + weights
  cvt_f32_bf16<<<(MTOK * HID) / 1024, 256, 0, stream>>>(X,  Xb,  MTOK * HID);
  cvt_f32_bf16<<<(HID * HID) / 1024, 256, 0, stream>>>(Wq, Wqb, HID * HID);
  cvt_f32_bf16<<<(HID * HID) / 1024, 256, 0, stream>>>(Wk, Wkb, HID * HID);
  cvt_f32_bf16<<<(HID * HID) / 1024, 256, 0, stream>>>(Wv, Wvb, HID * HID);
  cvt_f32_bf16<<<(HID * HID) / 1024, 256, 0, stream>>>(Wo, Wob, HID * HID);
  rope_table<<<SEQ, HDIM, 0, stream>>>(cosT, sinT);

  // 2) QKV projections (WMMA); V stored transposed for the PV fragments
  dim3 gGemm(HID / 128, MTOK / 128);   // (16, 32)
  gemm_bf16_wmma<0><<<gGemm, 256, 0, stream>>>(Xb, Wqb, Qb);
  gemm_bf16_wmma<0><<<gGemm, 256, 0, stream>>>(Xb, Wkb, Kb);
  gemm_bf16_wmma<1><<<gGemm, 256, 0, stream>>>(Xb, Wvb, VTb);

  // 3) RoPE in-place on Q and K
  rope_apply<<<BATCH * NHEADS * SEQ, 64, 0, stream>>>(Qb, cosT, sinT);
  rope_apply<<<BATCH * NHEADS * SEQ, 64, 0, stream>>>(Kb, cosT, sinT);

  // 4) fused causal flash attention
  flash_attn_wmma<<<dim3(BATCH * NHEADS, SEQ / 64), 128, 0, stream>>>(
      Qb, Kb, VTb, Ab);

  // 5) output projection -> f32 d_out
  gemm_bf16_wmma<2><<<gGemm, 256, 0, stream>>>(Ab, Wob, (float*)d_out);
}
